// Network_47253230191425
// MI455X (gfx1250) — matrix operand order repaired and checked by reading, not compiled, not run on hardware
//
#include <hip/hip_runtime.h>
#include <hip/hip_bf16.h>

typedef _Float16 f16;
typedef __attribute__((ext_vector_type(16))) _Float16 v16h;
typedef __attribute__((ext_vector_type(8)))  _Float16 v8h;
typedef __attribute__((ext_vector_type(8)))  float    v8f;

#define NN      20000
#define DEG     16
#define HEADS   4
#define HID     64
#define NP1     40
#define HIDDEN  256
#define INF     128

// ---------- elementwise f32 -> f16 ----------
__global__ void k_f32_to_f16(const float* __restrict__ in, f16* __restrict__ out, int n) {
  int i = blockIdx.x * blockDim.x + threadIdx.x;
  if (i < n) out[i] = (f16)in[i];
}

// ---------- W[K,N] f32 -> Wt[N,K] f16 ----------
__global__ void k_transpose_f16(const float* __restrict__ W, f16* __restrict__ Wt, int K, int N) {
  int i = blockIdx.x * blockDim.x + threadIdx.x;
  if (i < K * N) {
    int n = i / K, k = i - n * K;
    Wt[i] = (f16)W[k * N + n];
  }
}

// ---------- C[M,N] = A[M,K] * Bt[N,K]^T (+bias)(+relu) ----------
// One wave computes a 16 x (16*TN) strip. k-loop unrolled 2x with disjoint
// register sets so the second set's loads stay in flight behind the first
// set's WMMAs (partial s_wait_loadcnt instead of full waits).
template <int TN>
__global__ void k_gemm_wmma(const f16* __restrict__ A, const f16* __restrict__ Bt,
                            const float* __restrict__ bias, float* __restrict__ C,
                            int M, int N, int K, int act) {
  int wave = (int)((blockIdx.x * (unsigned)blockDim.x + threadIdx.x) >> 5);
  int lane = threadIdx.x & 31;
  int ngroups = N / (16 * TN);
  int total   = (M >> 4) * ngroups;
  if (wave >= total) return;                  // wave-uniform; EXEC stays all-1 for WMMA
  int mt = wave / ngroups, ng = wave - mt * ngroups;
  int l15 = lane & 15, hi = lane >> 4;

  const f16* Arow = A + (size_t)(mt * 16 + l15) * K;   // A row m, K-major
  const f16* Brow[TN];
#pragma unroll
  for (int t = 0; t < TN; ++t)
    Brow[t] = Bt + (size_t)(ng * 16 * TN + t * 16 + l15) * K;

  v8f acc[TN] = {};
  for (int k0 = 0; k0 < K; k0 += 64) {       // K is always a multiple of 64
    __builtin_prefetch(Arow + k0 + 256, 0, 3);  // global_prefetch_b8 into near caches
    // A layout (matches B): elem i -> K = i + hi*16  => two 16B chunks
    v8h a0 = *(const v8h*)(Arow + k0 + hi * 16);
    v8h a1 = *(const v8h*)(Arow + k0 + 8 + hi * 16);
    v8h c0 = *(const v8h*)(Arow + k0 + 32 + hi * 16);
    v8h c1 = *(const v8h*)(Arow + k0 + 40 + hi * 16);
    // B layout (matches A): elem i -> K = i + hi*16  => one 32B chunk per tile
    v16h b0[TN], b1[TN];
#pragma unroll
    for (int t = 0; t < TN; ++t) b0[t] = *(const v16h*)(Brow[t] + k0 + hi * 16);
#pragma unroll
    for (int t = 0; t < TN; ++t) b1[t] = *(const v16h*)(Brow[t] + k0 + 32 + hi * 16);

    v16h aa = __builtin_shufflevector(a0, a1, 0,1,2,3,4,5,6,7,8,9,10,11,12,13,14,15);
    v16h cc = __builtin_shufflevector(c0, c1, 0,1,2,3,4,5,6,7,8,9,10,11,12,13,14,15);
#pragma unroll
    for (int t = 0; t < TN; ++t)
      acc[t] = __builtin_amdgcn_wmma_f32_16x16x32_f16(false, aa, false, b0[t],
                                                      (short)0, acc[t], false, false);
#pragma unroll
    for (int t = 0; t < TN; ++t)
      acc[t] = __builtin_amdgcn_wmma_f32_16x16x32_f16(false, cc, false, b1[t],
                                                      (short)0, acc[t], false, false);
  }
#pragma unroll
  for (int t = 0; t < TN; ++t) {
    int col = ng * 16 * TN + t * 16 + l15;
    float bs = bias ? bias[col] : 0.0f;
    float* Cp = C + (size_t)(mt * 16 + hi * 8) * N + col;  // C/D: VGPR r -> row r + hi*8
    for (int r = 0; r < 8; ++r) {
      float v = acc[t][r] + bs;
      if (act == 1) v = fmaxf(v, 0.0f);
      Cp[(size_t)r * N] = v;
    }
  }
}

// ---------- el/er: per (node, head) dot of ht row with attention vectors ----------
__global__ void k_el_er(const float* __restrict__ ht, const float* __restrict__ al,
                        const float* __restrict__ ar, float* __restrict__ el,
                        float* __restrict__ er, int n_nodes, int H, int F) {
  int idx = blockIdx.x * blockDim.x + threadIdx.x;
  if (idx >= n_nodes * H) return;
  int n = idx / H, h = idx - n * H;
  const float* row = ht + ((size_t)n * H + h) * F;
  const float* pl  = al + h * F;
  const float* pr  = ar + h * F;
  float sl = 0.f, sr = 0.f;
  for (int f = 0; f < F; ++f) { float v = row[f]; sl += v * pl[f]; sr += v * pr[f]; }
  el[idx] = sl;
  er[idx] = sr;
}

// ---------- edge softmax + aggregation: one wave per (node, head), DEG=16 ----------
__global__ void k_attn_agg(const float* __restrict__ ht, const float* __restrict__ el,
                           const float* __restrict__ er, const int* __restrict__ src,
                           float* __restrict__ out, int n_nodes, int H, int F, int elu) {
  int gw   = (int)((blockIdx.x * (unsigned)blockDim.x + threadIdx.x) >> 5);
  int lane = threadIdx.x & 31;
  if (gw >= n_nodes * H) return;
  int n = gw / H, h = gw - n * H;

  int   s = 0;
  float e = -1e30f;
  if (lane < DEG) {
    s = src[n * DEG + lane];
    float v = el[s * H + h] + er[n * H + h];
    e = (v > 0.f) ? v : 0.2f * v;               // leaky_relu(., 0.2)
  }
  // wave max (lanes >= 16 hold -inf sentinel)
  float m = e;
  m = fmaxf(m, __shfl_xor(m, 1, 32));
  m = fmaxf(m, __shfl_xor(m, 2, 32));
  m = fmaxf(m, __shfl_xor(m, 4, 32));
  m = fmaxf(m, __shfl_xor(m, 8, 32));
  m = fmaxf(m, __shfl_xor(m, 16, 32));
  float ee = (lane < DEG) ? __expf(e - m) : 0.f;
  float d = ee;
  d += __shfl_xor(d, 1, 32);
  d += __shfl_xor(d, 2, 32);
  d += __shfl_xor(d, 4, 32);
  d += __shfl_xor(d, 8, 32);
  d += __shfl_xor(d, 16, 32);
  float inv = 1.0f / d;

  float acc0 = 0.f, acc1 = 0.f;
  for (int j = 0; j < DEG; ++j) {
    float a  = __shfl(ee, j, 32) * inv;
    int   sj = __shfl(s,  j, 32);
    const float* row = ht + ((size_t)sj * H + h) * F;
    if (lane < F)       acc0 += a * row[lane];
    if (lane + 32 < F)  acc1 += a * row[lane + 32];
  }
  float* orow = out + ((size_t)n * H + h) * F;
  if (lane < F) {
    float v = acc0;
    if (elu) v = (v > 0.f) ? v : (__expf(v) - 1.f);
    orow[lane] = v;
  }
  if (lane + 32 < F) {
    float v = acc1;
    if (elu) v = (v > 0.f) ? v : (__expf(v) - 1.f);
    orow[lane + 32] = v;
  }
}

// ---------- mean over heads: logits[n,p] = mean_h outc[n,h,p] ----------
__global__ void k_head_mean(const float* __restrict__ outc, float* __restrict__ logits,
                            int n_nodes, int H, int P) {
  int idx = blockIdx.x * blockDim.x + threadIdx.x;
  if (idx >= n_nodes * P) return;
  int n = idx / P, p = idx - n * P;
  float sacc = 0.f;
  for (int h = 0; h < H; ++h) sacc += outc[((size_t)n * H + h) * P + p];
  logits[idx] = sacc * (1.0f / H);
}

// ---------- dom = sigmoid(z1 @ d2_w + b): one wave per node, K=128 ----------
__global__ void k_dom(const float* __restrict__ z1, const float* __restrict__ w,
                      const float* __restrict__ b, float* __restrict__ dom, int n_nodes) {
  int gw   = (int)((blockIdx.x * (unsigned)blockDim.x + threadIdx.x) >> 5);
  int lane = threadIdx.x & 31;
  if (gw >= n_nodes) return;
  const float* row = z1 + (size_t)gw * 128;
  float sacc = row[lane] * w[lane] + row[lane + 32] * w[lane + 32]
             + row[lane + 64] * w[lane + 64] + row[lane + 96] * w[lane + 96];
  sacc += __shfl_xor(sacc, 1, 32);
  sacc += __shfl_xor(sacc, 2, 32);
  sacc += __shfl_xor(sacc, 4, 32);
  sacc += __shfl_xor(sacc, 8, 32);
  sacc += __shfl_xor(sacc, 16, 32);
  if (lane == 0) dom[gw] = 1.0f / (1.0f + __expf(-(sacc + b[0])));
}

extern "C" void kernel_launch(void* const* d_in, const int* in_sizes, int n_in,
                              void* d_out, int out_size, void* d_ws, size_t ws_size,
                              hipStream_t stream) {
  (void)in_sizes; (void)n_in; (void)out_size; (void)ws_size;
  const float* x    = (const float*)d_in[0];
  const int*   src  = (const int*)d_in[1];
  /* d_in[2] = dst is structurally repeat(arange(N), DEG) -> implicit */
  const float* W1   = (const float*)d_in[3];
  const float* al1  = (const float*)d_in[4];
  const float* ar1  = (const float*)d_in[5];
  const float* W2   = (const float*)d_in[6];
  const float* al2  = (const float*)d_in[7];
  const float* ar2  = (const float*)d_in[8];
  const float* Wc   = (const float*)d_in[9];
  const float* alc  = (const float*)d_in[10];
  const float* arc  = (const float*)d_in[11];
  const float* d0_w = (const float*)d_in[12];
  const float* d0_b = (const float*)d_in[13];
  const float* d1_w = (const float*)d_in[14];
  const float* d1_b = (const float*)d_in[15];
  const float* d2_w = (const float*)d_in[16];
  const float* d2_b = (const float*)d_in[17];

  float* out_h      = (float*)d_out;                       // [20000,256]
  float* out_logits = out_h + (size_t)NN * HIDDEN;         // [20000,40]
  float* out_dom    = out_logits + (size_t)NN * NP1;       // [20000]

  // ---- workspace carve-out (~62 MB, 256B aligned) ----
  size_t off = 0;
  char* ws = (char*)d_ws;
  auto wsa = [&](size_t bytes) -> void* {
    void* p = ws + off;
    off = (off + bytes + 255) & ~(size_t)255;
    return p;
  };
  float* bufA = (float*)wsa((size_t)NN * HIDDEN * 4);   // GEMM outputs (ht)
  float* bufB = (float*)wsa((size_t)NN * HIDDEN * 4);   // agg outputs / z0
  f16*   bufC = (f16*)  wsa((size_t)NN * HIDDEN * 2);   // f16 activations
  f16*   xh   = (f16*)  wsa((size_t)NN * INF * 2);      // x in f16
  f16*   zh   = (f16*)  wsa((size_t)NN * 128 * 2);      // z0 in f16
  float* elb  = (float*)wsa((size_t)NN * HEADS * 4);
  float* erb  = (float*)wsa((size_t)NN * HEADS * 4);
  f16*   w1t  = (f16*)wsa((size_t)HIDDEN * INF * 2);    // [256,128]
  f16*   w2t  = (f16*)wsa((size_t)HIDDEN * HIDDEN * 2); // [256,256]
  f16*   wct  = (f16*)wsa((size_t)(HEADS*NP1) * HIDDEN * 2); // [160,256]
  f16*   d0t  = (f16*)wsa((size_t)128 * HIDDEN * 2);    // [128,256]
  f16*   d1t  = (f16*)wsa((size_t)128 * 128 * 2);       // [128,128]

  auto cdiv = [](int a, int b) { return (a + b - 1) / b; };
  auto gemm = [&](const f16* A, const f16* Bt, const float* bias, float* C,
                  int M, int N, int K, int act) {
    if (N % 64 == 0) {
      int groups = (M / 16) * (N / 64);
      k_gemm_wmma<4><<<cdiv(groups, 8), 256, 0, stream>>>(A, Bt, bias, C, M, N, K, act);
    } else {
      int groups = (M / 16) * (N / 32);
      k_gemm_wmma<2><<<cdiv(groups, 8), 256, 0, stream>>>(A, Bt, bias, C, M, N, K, act);
    }
  };

  // ---- prep: f16 conversions + weight transposes ----
  k_f32_to_f16<<<cdiv(NN * INF, 256), 256, 0, stream>>>(x, xh, NN * INF);
  k_transpose_f16<<<cdiv(INF * HIDDEN, 256), 256, 0, stream>>>(W1, w1t, INF, HIDDEN);
  k_transpose_f16<<<cdiv(HIDDEN * HIDDEN, 256), 256, 0, stream>>>(W2, w2t, HIDDEN, HIDDEN);
  k_transpose_f16<<<cdiv(HIDDEN * HEADS * NP1, 256), 256, 0, stream>>>(Wc, wct, HIDDEN, HEADS * NP1);
  k_transpose_f16<<<cdiv(HIDDEN * 128, 256), 256, 0, stream>>>(d0_w, d0t, HIDDEN, 128);
  k_transpose_f16<<<cdiv(128 * 128, 256), 256, 0, stream>>>(d1_w, d1t, 128, 128);

  const int nh_blocks  = cdiv(NN * HEADS, 256);       // threads for el/er
  const int agg_blocks = cdiv(NN * HEADS, 8);         // 8 waves per 256-thread block

  // ---- GAT layer 1: 128 -> 4x64, ELU ----
  gemm(xh, w1t, nullptr, bufA, NN, HIDDEN, INF, 0);
  k_el_er<<<nh_blocks, 256, 0, stream>>>(bufA, al1, ar1, elb, erb, NN, HEADS, HID);
  k_attn_agg<<<agg_blocks, 256, 0, stream>>>(bufA, elb, erb, src, bufB, NN, HEADS, HID, 1);
  k_f32_to_f16<<<cdiv(NN * HIDDEN, 256), 256, 0, stream>>>(bufB, bufC, NN * HIDDEN);

  // ---- GAT layer 2: 256 -> 4x64, ELU; h2 written directly into d_out ----
  gemm(bufC, w2t, nullptr, bufA, NN, HIDDEN, HIDDEN, 0);
  k_el_er<<<nh_blocks, 256, 0, stream>>>(bufA, al2, ar2, elb, erb, NN, HEADS, HID);
  k_attn_agg<<<agg_blocks, 256, 0, stream>>>(bufA, elb, erb, src, out_h, NN, HEADS, HID, 1);
  k_f32_to_f16<<<cdiv(NN * HIDDEN, 256), 256, 0, stream>>>(out_h, bufC, NN * HIDDEN);

  // ---- classifier GATConv: 256 -> 4x40 (N=160 -> TN=2), then mean over heads ----
  gemm(bufC, wct, nullptr, bufA, NN, HEADS * NP1, HIDDEN, 0);
  k_el_er<<<nh_blocks, 256, 0, stream>>>(bufA, alc, arc, elb, erb, NN, HEADS, NP1);
  k_attn_agg<<<agg_blocks, 256, 0, stream>>>(bufA, elb, erb, src, bufB, NN, HEADS, NP1, 0);
  k_head_mean<<<cdiv(NN * NP1, 256), 256, 0, stream>>>(bufB, out_logits, NN, HEADS, NP1);

  // ---- discriminator MLP: 256 -> 128 (relu) -> 128 (relu) -> 1 (sigmoid) ----
  gemm(bufC, d0t, d0_b, bufB, NN, 128, HIDDEN, 1);                       // z0 = relu(h@d0_w+b)
  k_f32_to_f16<<<cdiv(NN * 128, 256), 256, 0, stream>>>(bufB, zh, NN * 128);
  gemm(zh, d1t, d1_b, bufA, NN, 128, 128, 1);                            // z1 = relu(z0@d1_w+b)
  k_dom<<<cdiv(NN, 8), 256, 0, stream>>>(bufA, d2_w, d2_b, out_dom, NN); // sigmoid(z1@d2_w+b)
}
